// PropositionVQVAE_27668179321221
// MI455X (gfx1250) — matrix operand order repaired and checked by reading, not compiled, not run on hardware
//
#include <hip/hip_runtime.h>

// ---------------- problem constants ----------------
#define B_ROWS 16384
#define E_DIM  256
#define H_DIM  1024
#define C_DIM  256
#define K_CB   16384
#define X_DIM  768   // 3*E

typedef unsigned short ushort_t;
typedef unsigned int   u32;
typedef unsigned long long u64;

typedef __attribute__((ext_vector_type(16))) __bf16 v16bf;
typedef __attribute__((ext_vector_type(8)))  float  v8f;

// ---- CDNA5 async global->LDS path (guarded; falls back to reg staging) ----
#if defined(__HIP_DEVICE_COMPILE__) && defined(__has_builtin)
#if __has_builtin(__builtin_amdgcn_global_load_async_to_lds_b128)
#define USE_ASYNC_LDS 1
#endif
#endif

#if defined(USE_ASYNC_LDS)
// builtin signature (from hipcc diagnostic): (v4i as1*, v4i as3*, imm, imm)
typedef int v4i_vs __attribute__((vector_size(16)));
typedef __attribute__((address_space(1))) v4i_vs* as1_v4i;
typedef __attribute__((address_space(3))) v4i_vs* as3_v4i;
__device__ __forceinline__ void async_copy16(const ushort_t* g, ushort_t* l) {
  __builtin_amdgcn_global_load_async_to_lds_b128(
      (as1_v4i)g, (as3_v4i)l, 0, 0);
}
#if __has_builtin(__builtin_amdgcn_s_wait_asynccnt)
#define WAIT_ASYNC() __builtin_amdgcn_s_wait_asynccnt(0)
#else
#define WAIT_ASYNC() asm volatile("s_wait_asynccnt 0x0" ::: "memory")
#endif
#endif

// fp32 -> bf16, round to nearest even (storage is ushort)
__device__ __forceinline__ ushort_t f2bf(float f) {
  u32 x = __float_as_uint(f);
  u32 r = x + 0x7FFFu + ((x >> 16) & 1u);
  return (ushort_t)(r >> 16);
}

// monotonic float->u32 mapping (order-preserving)
__device__ __forceinline__ u32 mono_f32(float f) {
  u32 b = __float_as_uint(f);
  return (b & 0x80000000u) ? ~b : (b | 0x80000000u);
}
// pack (score, col): max over packed == max score, ties -> smallest col
__device__ __forceinline__ u64 pack_score(float s, int col) {
  return ((u64)mono_f32(s) << 32) | (u64)(0xFFFFFFFFu - (u32)col);
}

// ---------------- bf16 WMMA GEMM ----------------
// C[m,n] = sum_k A[m,k] * Bt[n,k]   (A: MxK row-major bf16, Bt: NxK row-major bf16)
// EPI 0: out_bf = bf16(relu(acc + bias))
// EPI 1: out_bf = bf16(acc + bias), out_f32 = acc + bias        (z layer)
// EPI 2: row argmax of (acc + biasScale*bias[col]) -> atomicMax best[row]
// EPI 3: split-3 f32 scatter: col n -> out_f32[(n>>8)*M*256 + row*256 + (n&255)]
#define BM 128
#define BN 256
#define BK 32
#define LDSP 40   // padded LDS row stride (elements) to spread banks

union FragBf { v16bf v; uint4 q[2]; };

template <int EPI>
__global__ __launch_bounds__(256) void gemm_bf16_wmma(
    const ushort_t* __restrict__ A, const ushort_t* __restrict__ Bt,
    const float* __restrict__ bias, float biasScale,
    ushort_t* __restrict__ outBf, float* __restrict__ outF32,
    u64* __restrict__ best, int M, int N, int K)
{
  // double-buffered tiles: A 128x32, Bt 256x32 (bf16), padded stride 40
  __shared__ __align__(16) ushort_t lA[2 * BM * LDSP];
  __shared__ __align__(16) ushort_t lB[2 * BN * LDSP];

  const int tid   = threadIdx.x;
  const int wid   = tid >> 5;
  const int lane  = tid & 31;
  const int half  = lane >> 4;
  const int l15   = lane & 15;
  const int waveM = wid >> 2;    // 0..1 -> 64 rows each
  const int waveN = wid & 3;     // 0..3 -> 64 cols each
  const int bm = blockIdx.y * BM;
  const int bn = blockIdx.x * BN;

  // per-thread staging geometry: A tile = 512 16B-chunks, B tile = 1024
  int aRow[2], aCo[2], bRow[4], bCo[4];
#pragma unroll
  for (int i = 0; i < 2; ++i) {
    int c = tid + i * 256; aRow[i] = c >> 2; aCo[i] = (c & 3) << 3;
  }
#pragma unroll
  for (int i = 0; i < 4; ++i) {
    int c = tid + i * 256; bRow[i] = c >> 2; bCo[i] = (c & 3) << 3;
  }

#if defined(USE_ASYNC_LDS)
  auto stageAsync = [&](int k0, int buf) {
#pragma unroll
    for (int i = 0; i < 2; ++i)
      async_copy16(A + (size_t)(bm + aRow[i]) * K + (k0 + aCo[i]),
                   &lA[buf * (BM * LDSP) + aRow[i] * LDSP + aCo[i]]);
#pragma unroll
    for (int i = 0; i < 4; ++i)
      async_copy16(Bt + (size_t)(bn + bRow[i]) * K + (k0 + bCo[i]),
                   &lB[buf * (BN * LDSP) + bRow[i] * LDSP + bCo[i]]);
  };
#else
  auto loadTile = [&](int k0, uint4* ra, uint4* rb) {
#pragma unroll
    for (int i = 0; i < 2; ++i)
      ra[i] = *(const uint4*)(A + (size_t)(bm + aRow[i]) * K + (k0 + aCo[i]));
#pragma unroll
    for (int i = 0; i < 4; ++i)
      rb[i] = *(const uint4*)(Bt + (size_t)(bn + bRow[i]) * K + (k0 + bCo[i]));
  };
  auto storeTile = [&](int buf, const uint4* ra, const uint4* rb) {
#pragma unroll
    for (int i = 0; i < 2; ++i)
      *(uint4*)&lA[buf * (BM * LDSP) + aRow[i] * LDSP + aCo[i]] = ra[i];
#pragma unroll
    for (int i = 0; i < 4; ++i)
      *(uint4*)&lB[buf * (BN * LDSP) + bRow[i] * LDSP + bCo[i]] = rb[i];
  };
#endif

  const v8f vzero = {0.f,0.f,0.f,0.f,0.f,0.f,0.f,0.f};
  v8f acc[4][4];
#pragma unroll
  for (int mi = 0; mi < 4; ++mi)
#pragma unroll
    for (int ni = 0; ni < 4; ++ni) acc[mi][ni] = vzero;

  const int ntiles = K / BK;

  // prologue: stage tile 0 into buffer 0
#if defined(USE_ASYNC_LDS)
  stageAsync(0, 0);
  WAIT_ASYNC();
#else
  {
    uint4 ra[2], rb[4];
    loadTile(0, ra, rb);
    storeTile(0, ra, rb);
  }
#endif
  __syncthreads();

  for (int t = 0; t < ntiles; ++t) {
    const int cur = t & 1;
#if defined(USE_ASYNC_LDS)
    if (t + 1 < ntiles) stageAsync((t + 1) * BK, cur ^ 1);  // overlaps WMMAs
#else
    uint4 ra[2], rb[4];
    if (t + 1 < ntiles) loadTile((t + 1) * BK, ra, rb);
#endif

    // per-lane fragment gathers: contiguous b128 LDS runs per the ISA layout
    const ushort_t* la = &lA[cur * (BM * LDSP)];
    const ushort_t* lb = &lB[cur * (BN * LDSP)];
    FragBf fa[4], fb[4];
#pragma unroll
    for (int mi = 0; mi < 4; ++mi) {
      int r = waveM * 64 + mi * 16 + l15;
      fa[mi].q[0] = *(const uint4*)&la[r * LDSP + half * 8];       // K = h*8..
      fa[mi].q[1] = *(const uint4*)&la[r * LDSP + half * 8 + 16];  // K = 16+h*8..
    }
#pragma unroll
    for (int ni = 0; ni < 4; ++ni) {
      int r = waveN * 64 + ni * 16 + l15;
      fb[ni].q[0] = *(const uint4*)&lb[r * LDSP + half * 8];       // K = h*8..
      fb[ni].q[1] = *(const uint4*)&lb[r * LDSP + half * 8 + 16];
    }

#pragma unroll
    for (int mi = 0; mi < 4; ++mi)
#pragma unroll
      for (int ni = 0; ni < 4; ++ni)
        acc[mi][ni] = __builtin_amdgcn_wmma_f32_16x16x32_bf16(
            false, fa[mi].v, false, fb[ni].v, (short)0, acc[mi][ni], false, false);

#if defined(USE_ASYNC_LDS)
    if (t + 1 < ntiles) WAIT_ASYNC();
#else
    if (t + 1 < ntiles) storeTile(cur ^ 1, ra, rb);
#endif
    __syncthreads();
  }

  if (EPI == 2) {
    // fused row-wise argmax of score = acc - 0.5*|c|^2 (biasScale = -0.5)
    u64* lred = (u64*)lA;  // [128 rows][4 waveN]; loop-exit barrier protects reuse
#pragma unroll
    for (int mi = 0; mi < 4; ++mi) {
#pragma unroll
      for (int r = 0; r < 8; ++r) {
        float s = -3.4e38f; int c = 0;
#pragma unroll
        for (int ni = 0; ni < 4; ++ni) {
          int   cc = bn + waveN * 64 + ni * 16 + l15;
          float ss = acc[mi][ni][r] + biasScale * bias[cc];
          if (ss > s || (ss == s && cc < c)) { s = ss; c = cc; }
        }
#pragma unroll
        for (int m = 1; m < 16; m <<= 1) {
          float os = __shfl_xor(s, m, 32);
          int   oc = __shfl_xor(c, m, 32);
          if (os > s || (os == s && oc < c)) { s = os; c = oc; }
        }
        if (l15 == 0) {
          int rl = waveM * 64 + mi * 16 + half * 8 + r;
          lred[rl * 4 + waveN] = pack_score(s, c);
        }
      }
    }
    __syncthreads();
    if (tid < BM) {
      u64 v = lred[tid * 4];
#pragma unroll
      for (int j = 1; j < 4; ++j) { u64 w = lred[tid * 4 + j]; if (w > v) v = w; }
      atomicMax(&best[bm + tid], v);
    }
    return;
  }

  // standard epilogues
#pragma unroll
  for (int mi = 0; mi < 4; ++mi) {
#pragma unroll
    for (int ni = 0; ni < 4; ++ni) {
      int   col = bn + waveN * 64 + ni * 16 + l15;
      float bv  = bias ? biasScale * bias[col] : 0.f;
#pragma unroll
      for (int r = 0; r < 8; ++r) {
        int   row = bm + waveM * 64 + mi * 16 + half * 8 + r;
        float v   = acc[mi][ni][r] + bv;
        if (EPI == 0) {
          v = v > 0.f ? v : 0.f;
          outBf[(size_t)row * N + col] = f2bf(v);
        } else if (EPI == 1) {
          outBf [(size_t)row * N + col] = f2bf(v);
          outF32[(size_t)row * N + col] = v;
        } else { // EPI == 3
          int sec = col >> 8, e = col & 255;
          outF32[(size_t)sec * (size_t)M * 256 + (size_t)row * 256 + e] = v;
        }
      }
    }
  }
}

// ---------------- glue kernels ----------------
__global__ void build_x(const float* __restrict__ s, const float* __restrict__ r,
                        const float* __restrict__ o, ushort_t* __restrict__ X)
{
  size_t idx = (size_t)blockIdx.x * 256 + threadIdx.x;
  int row = (int)(idx / X_DIM), j = (int)(idx % X_DIM);
  float v = (j < 256) ? s[(size_t)row * 256 + j]
          : (j < 512) ? r[(size_t)row * 256 + (j - 256)]
                      : o[(size_t)row * 256 + (j - 512)];
  X[idx] = f2bf(v);
}

__global__ void tconv(const float* __restrict__ W, ushort_t* __restrict__ Wt,
                      int K, int N)
{
  size_t idx = (size_t)blockIdx.x * 256 + threadIdx.x;
  if (idx >= (size_t)K * N) return;
  int k = (int)(idx / N), n = (int)(idx % N);
  Wt[(size_t)n * K + k] = f2bf(W[idx]);
}

__global__ void conv_bf16(const float* __restrict__ src, ushort_t* __restrict__ dst,
                          size_t n)
{
  size_t i = (size_t)blockIdx.x * 256 + threadIdx.x;
  if (i < n) dst[i] = f2bf(src[i]);
}

__global__ void row_sq(const float* __restrict__ cb, float* __restrict__ out)
{
  int wid = threadIdx.x >> 5, lane = threadIdx.x & 31;
  int row = blockIdx.x * 8 + wid;
  float s = 0.f;
#pragma unroll
  for (int j = 0; j < C_DIM / 32; ++j) {
    float v = cb[(size_t)row * C_DIM + lane + 32 * j];
    s += v * v;
  }
#pragma unroll
  for (int m = 16; m >= 1; m >>= 1) s += __shfl_xor(s, m, 32);
  if (lane == 0) out[row] = s;
}

__global__ void init_state(u64* __restrict__ best, double* __restrict__ lossSum)
{
  size_t i = (size_t)blockIdx.x * 256 + threadIdx.x;
  if (i < B_ROWS) best[i] = 0ull;
  if (i == 0) *lossSum = 0.0;
}

__global__ void vq_post(const u64* __restrict__ best, const float* __restrict__ cb,
                        const float* __restrict__ z, ushort_t* __restrict__ zq,
                        float* __restrict__ codes_out, double* __restrict__ lossSum)
{
  __shared__ int   sc;
  __shared__ float wsum[8];
  int row = blockIdx.x, tid = threadIdx.x;
  if (tid == 0) {
    u32 low = (u32)(best[row] & 0xFFFFFFFFull);
    sc = (int)(0xFFFFFFFFu - low);
    codes_out[row] = (float)sc;
  }
  __syncthreads();
  int   code = sc;
  float cv = cb[(size_t)code * C_DIM + tid];
  float zv = z [(size_t)row  * C_DIM + tid];
  zq[(size_t)row * C_DIM + tid] = f2bf(cv);
  float d = cv - zv, s = d * d;
#pragma unroll
  for (int m = 16; m >= 1; m >>= 1) s += __shfl_xor(s, m, 32);
  if ((tid & 31) == 0) wsum[tid >> 5] = s;
  __syncthreads();
  if (tid == 0) {
    float t = 0.f;
#pragma unroll
    for (int i = 0; i < 8; ++i) t += wsum[i];
    atomicAdd(lossSum, (double)t);
  }
}

__global__ void finalize_loss(const double* __restrict__ lossSum,
                              float* __restrict__ out)
{
  float mean = (float)(*lossSum / ((double)B_ROWS * (double)C_DIM));
  out[0] = mean;           // codebook_loss
  out[1] = mean * 0.25f;   // commitment_loss (BETA)
  out[2] = mean * 1.25f;   // total_vq
}

// ---------------- host launch ----------------
extern "C" void kernel_launch(void* const* d_in, const int* in_sizes, int n_in,
                              void* d_out, int out_size, void* d_ws, size_t ws_size,
                              hipStream_t stream) {
  (void)in_sizes; (void)n_in; (void)out_size; (void)ws_size;
  const float* subj  = (const float*)d_in[0];
  const float* rel   = (const float*)d_in[1];
  const float* obj   = (const float*)d_in[2];
  const float* ew1   = (const float*)d_in[3];
  const float* eb1   = (const float*)d_in[4];
  const float* ew2   = (const float*)d_in[5];
  const float* eb2   = (const float*)d_in[6];
  const float* ew3   = (const float*)d_in[7];
  const float* eb3   = (const float*)d_in[8];
  const float* cb    = (const float*)d_in[9];
  const float* dw1   = (const float*)d_in[10];
  const float* db1   = (const float*)d_in[11];
  const float* dw2   = (const float*)d_in[12];
  const float* db2   = (const float*)d_in[13];
  const float* dw3   = (const float*)d_in[14];
  const float* db3   = (const float*)d_in[15];
  float* out = (float*)d_out;

  char* ws = (char*)d_ws; size_t off = 0;
  auto carve = [&](size_t bytes) -> void* {
    void* p = ws + off; off = (off + bytes + 255) & ~(size_t)255; return p;
  };
  ushort_t* actA  = (ushort_t*)carve((size_t)B_ROWS * H_DIM * 2); // X, h2, d1
  ushort_t* actB  = (ushort_t*)carve((size_t)B_ROWS * H_DIM * 2); // h1, d2
  ushort_t* w1t   = (ushort_t*)carve((size_t)H_DIM * X_DIM * 2);
  ushort_t* w2t   = (ushort_t*)carve((size_t)H_DIM * H_DIM * 2);
  ushort_t* w3t   = (ushort_t*)carve((size_t)C_DIM * H_DIM * 2);
  ushort_t* dw1t  = (ushort_t*)carve((size_t)H_DIM * C_DIM * 2);
  ushort_t* dw2t  = (ushort_t*)carve((size_t)H_DIM * H_DIM * 2);
  ushort_t* dw3t  = (ushort_t*)carve((size_t)X_DIM * H_DIM * 2);
  ushort_t* cbbf  = (ushort_t*)carve((size_t)K_CB * C_DIM * 2);
  float*    zf    = (float*)   carve((size_t)B_ROWS * C_DIM * 4);
  ushort_t* zbf   = (ushort_t*)carve((size_t)B_ROWS * C_DIM * 2);
  ushort_t* zqbf  = (ushort_t*)carve((size_t)B_ROWS * C_DIM * 2);
  float*    cbsq  = (float*)   carve((size_t)K_CB * 4);
  u64*      best  = (u64*)     carve((size_t)B_ROWS * 8);
  double*   lsum  = (double*)  carve(256);

  float* codes_out = out + (size_t)3 * B_ROWS * E_DIM;
  float* loss_out  = codes_out + B_ROWS;

  // preprocessing
  build_x<<<(B_ROWS * X_DIM) / 256, 256, 0, stream>>>(subj, rel, obj, actA);
  tconv<<<(X_DIM * H_DIM + 255) / 256, 256, 0, stream>>>(ew1, w1t, X_DIM, H_DIM);
  tconv<<<(H_DIM * H_DIM + 255) / 256, 256, 0, stream>>>(ew2, w2t, H_DIM, H_DIM);
  tconv<<<(H_DIM * C_DIM + 255) / 256, 256, 0, stream>>>(ew3, w3t, H_DIM, C_DIM);
  tconv<<<(C_DIM * H_DIM + 255) / 256, 256, 0, stream>>>(dw1, dw1t, C_DIM, H_DIM);
  tconv<<<(H_DIM * H_DIM + 255) / 256, 256, 0, stream>>>(dw2, dw2t, H_DIM, H_DIM);
  tconv<<<(H_DIM * X_DIM + 255) / 256, 256, 0, stream>>>(dw3, dw3t, H_DIM, X_DIM);
  conv_bf16<<<((size_t)K_CB * C_DIM + 255) / 256, 256, 0, stream>>>(
      cb, cbbf, (size_t)K_CB * C_DIM);
  row_sq<<<K_CB / 8, 256, 0, stream>>>(cb, cbsq);
  init_state<<<(B_ROWS + 255) / 256, 256, 0, stream>>>(best, lsum);

  // encoder
  gemm_bf16_wmma<0><<<dim3(H_DIM / BN, B_ROWS / BM), 256, 0, stream>>>(
      actA, w1t, eb1, 1.f, actB, nullptr, nullptr, B_ROWS, H_DIM, X_DIM);   // h1
  gemm_bf16_wmma<0><<<dim3(H_DIM / BN, B_ROWS / BM), 256, 0, stream>>>(
      actB, w2t, eb2, 1.f, actA, nullptr, nullptr, B_ROWS, H_DIM, H_DIM);   // h2
  gemm_bf16_wmma<1><<<dim3(C_DIM / BN, B_ROWS / BM), 256, 0, stream>>>(
      actA, w3t, eb3, 1.f, zbf, zf, nullptr, B_ROWS, C_DIM, H_DIM);         // z

  // VQ: fused scores + argmax (score = z.c - 0.5|c|^2)
  gemm_bf16_wmma<2><<<dim3(K_CB / BN, B_ROWS / BM), 256, 0, stream>>>(
      zbf, cbbf, cbsq, -0.5f, nullptr, nullptr, best, B_ROWS, K_CB, C_DIM);
  vq_post<<<B_ROWS, 256, 0, stream>>>(best, cb, zf, zqbf, codes_out, lsum);
  finalize_loss<<<1, 1, 0, stream>>>(lsum, loss_out);

  // decoder
  gemm_bf16_wmma<0><<<dim3(H_DIM / BN, B_ROWS / BM), 256, 0, stream>>>(
      zqbf, dw1t, db1, 1.f, actA, nullptr, nullptr, B_ROWS, H_DIM, C_DIM);  // d1
  gemm_bf16_wmma<0><<<dim3(H_DIM / BN, B_ROWS / BM), 256, 0, stream>>>(
      actA, dw2t, db2, 1.f, actB, nullptr, nullptr, B_ROWS, H_DIM, H_DIM);  // d2
  gemm_bf16_wmma<3><<<dim3(X_DIM / BN, B_ROWS / BM), 256, 0, stream>>>(
      actB, dw3t, db3, 1.f, nullptr, out, nullptr, B_ROWS, X_DIM, H_DIM);   // split
}
